// SRPsiEngineTiny_88450556494483
// MI455X (gfx1250) — matrix operand order repaired and checked by reading, not compiled, not run on hardware
//
#include <hip/hip_runtime.h>

#define DEVINL __device__ __forceinline__

typedef __attribute__((ext_vector_type(16))) __bf16 v16bf;
typedef __attribute__((ext_vector_type(8)))  float  v8f;

constexpr int   cH    = 128;
constexpr int   cH2   = 256;
constexpr int   cK    = 5;
constexpr int   cDEPTH= 3;
constexpr int   cTOUT = 16;
constexpr int   cB    = 8;
constexpr int   cTIN  = 64;
constexpr int   cX    = 1024;
constexpr float cDT   = 0.01f;

union Frag { v16bf v; unsigned u[8]; };

// A-matrix 16x32 bf16 fragment (row-major tile, leading dim ld elements).
// lanes 0-15: M=lane, K-base 0 ; lanes 16-31: M=lane-16, K-base 8.
// VGPR j<4 -> K = kb+2j,+1 ; VGPR j>=4 -> K = 16+kb+2(j-4),+1
DEVINL void load_frag_a(const __bf16* base, int ld, Frag& f, int lane) {
  const int m  = lane & 15;
  const int kb = (lane >> 4) << 3;
  const __bf16* row = base + m * ld;
#pragma unroll
  for (int j = 0; j < 8; ++j) {
    const int k = (j < 4) ? (kb + 2 * j) : (16 + kb + 2 * (j - 4));
    f.u[j] = *(const unsigned*)(row + k);
  }
}

// B-matrix 32x16 bf16 fragment built from W^T: element(k,n) = W[n][k].
// lanes 0-15: K-base 0 ; lanes 16-31: K-base 8; VGPR j<4 -> K = kb+2j,+1 ; VGPR j>=4 -> K = 16+kb+2(j-4),+1
DEVINL void load_frag_b(const __bf16* w, int ldw, Frag& f, int lane) {
  const int n  = lane & 15;
  const int kb = (lane >> 4) << 3;
  const __bf16* row = w + n * ldw;
#pragma unroll
  for (int j = 0; j < 8; ++j) {
    const int k = (j < 4) ? (kb + 2 * j) : (16 + kb + 2 * (j - 4));
    f.u[j] = *(const unsigned*)(row + k);
  }
}

DEVINL float activate(float v, int act) {
  switch (act) {
    case 1: return 0.5f * v * (1.f + erff(v * 0.70710678118654752f)); // exact GELU
    case 2: return tanhf(v);
    case 3: return 1.f / (1.f + __expf(-v));                          // gate sigmoid
    default: return v;
  }
}

struct Seg { const __bf16* A; int lda; const __bf16* W; int ldw; int K; };

// M=64 GEMM, 8 wave32s: wave w -> row-tile (w&3), column half (w>>2).
// out = act( sum_segs A_s @ W_s^T + bias ), epilogue per C element.
template <class Epi>
DEVINL void gemm64(const Seg* segs, int nseg, int N, Epi epi) {
  const int tid  = (int)threadIdx.x;
  const int lane = tid & 31;
  const int wave = tid >> 5;
  const int m0   = (wave & 3) << 4;
  const int ch   = wave >> 2;
  const int halfN = N >> 1;
  for (int n0 = ch * halfN; n0 < ch * halfN + halfN; n0 += 16) {
    v8f acc = {0.f, 0.f, 0.f, 0.f, 0.f, 0.f, 0.f, 0.f};
    for (int s = 0; s < nseg; ++s) {
      const Seg sg = segs[s];
      __builtin_prefetch(sg.W + n0 * sg.ldw, 0, 2);   // warm W rows (global_prefetch_b8)
      for (int k0 = 0; k0 < sg.K; k0 += 32) {
        Frag a, b;
        load_frag_a(sg.A + m0 * sg.lda + k0, sg.lda, a, lane);
        load_frag_b(sg.W + n0 * sg.ldw + k0, sg.ldw, b, lane);
        acc = __builtin_amdgcn_wmma_f32_16x16x32_bf16(
            false, a.v, false, b.v, (short)0, acc, false, false);
      }
    }
    const int mm = m0 + ((lane >> 4) << 3);
    const int nn = n0 + (lane & 15);
#pragma unroll
    for (int r = 0; r < 8; ++r) epi(mm + r, nn, acc[r]);
  }
}

struct EpiBf {   // bias + act -> bf16 LDS (bias must be non-null)
  const float* bias; __bf16* out; int ldo; int act;
  DEVINL void operator()(int m, int n, float v) const {
    v += bias[n];
    out[m * ldo + n] = (__bf16)activate(v, act);
  }
};
struct EpiF32 {  // bias + act -> f32 LDS
  const float* bias; float* out; int ldo; int act;
  DEVINL void operator()(int m, int n, float v) const {
    v += bias[n];
    out[m * ldo + n] = activate(v, act);
  }
};
struct EpiEnc {  // encoder: bias, clip +-10, split re/im -> f32 global
  const float* bias; float* re; float* im; int ld;
  DEVINL void operator()(int m, int n, float v) const {
    v += bias[n];
    v = fminf(10.f, fmaxf(-10.f, v));
    if (n < cH) re[m * ld + n] = v;
    else        im[m * ld + (n - cH)] = v;
  }
};

// ---------------- prep kernels ----------------
__global__ void cvt_kernel(const float* __restrict__ in, __bf16* __restrict__ out, int n) {
  int i = blockIdx.x * 256 + threadIdx.x;
  if (i < n) out[i] = (__bf16)in[i];
}
// s_rw [D][O][I][K] f32 -> [D][K][O][I] bf16 so every tap is a contiguous HxH GEMM operand
__global__ void conv_repack_kernel(const float* __restrict__ in, __bf16* __restrict__ out, int n) {
  int i = blockIdx.x * 256 + threadIdx.x;
  if (i >= n) return;
  int k = i % cK; int rest = i / cK;
  int c = rest % cH; rest /= cH;
  int o = rest % cH; int d = rest / cH;
  out[(((size_t)d * cK + k) * cH + o) * cH + c] = (__bf16)in[i];
}

// ---------------- encoder ----------------
__global__ __launch_bounds__(256) void enc_kernel(
    const float* __restrict__ xin, const __bf16* __restrict__ w1, const float* __restrict__ b1,
    const __bf16* __restrict__ w2, const float* __restrict__ b2,
    float* __restrict__ oRe, float* __restrict__ oIm) {
  extern __shared__ char smem[];
  __bf16* sA = (__bf16*)smem;   // 64 x 64 normalized input (x-major)
  __bf16* sT = sA + 64 * 64;    // 64 x 128 hidden
  const int b  = (int)blockIdx.x >> 4;
  const int x0 = ((int)blockIdx.x & 15) << 6;
  const int tid = (int)threadIdx.x;
  // 4 threads per x-position; wave32 shuffle reduction for mean/std
  {
    const int m = tid >> 2;          // x within tile
    const int p = tid & 3;           // t quarter (16 samples)
    const int x = x0 + m;
    const float* col = xin + (size_t)b * cTIN * cX + x;
    float s = 0.f, sq = 0.f;
    for (int t = p * 16; t < p * 16 + 16; ++t) {
      const float v = col[(size_t)t * cX]; s += v; sq += v * v;
    }
    s  += __shfl_xor(s, 1);  sq += __shfl_xor(sq, 1);
    s  += __shfl_xor(s, 2);  sq += __shfl_xor(sq, 2);
    const float mean = s / cTIN;
    const float var  = (sq - cTIN * mean * mean) / (cTIN - 1);  // unbiased (torch std)
    const float inv  = 1.f / (sqrtf(fmaxf(var, 0.f)) + 1e-6f);
    for (int t = p * 16; t < p * 16 + 16; ++t)
      sA[m * 64 + t] = (__bf16)((col[(size_t)t * cX] - mean) * inv);
  }
  __syncthreads();
  { Seg s = { sA, 64, w1, cTIN, cTIN }; gemm64(&s, 1, cH, EpiBf{ b1, sT, cH, 1 }); }
  __syncthreads();
  { Seg s = { sT, cH, w2, cH, cH };
    gemm64(&s, 1, cH2, EpiEnc{ b2, oRe + ((size_t)b * cX + x0) * cH,
                                    oIm + ((size_t)b * cX + x0) * cH, cH }); }
}

// ---------------- one depth block (fused) ----------------
__global__ __launch_bounds__(256) void block_kernel(
    const float* __restrict__ pRe, const float* __restrict__ pIm,
    float* __restrict__ oRe, float* __restrict__ oIm,
    const __bf16* __restrict__ convR, const __bf16* __restrict__ convI,
    const __bf16* __restrict__ mW,  const __bf16* __restrict__ rW1,
    const __bf16* __restrict__ rW2, const __bf16* __restrict__ nW1,
    const __bf16* __restrict__ nW2, const __bf16* __restrict__ nGw,
    const float* __restrict__ s_rb, const float* __restrict__ s_ib,
    const float* __restrict__ s_mb, const float* __restrict__ r_b1,
    const float* __restrict__ r_b2, const float* __restrict__ n_b1,
    const float* __restrict__ n_b2, const float* __restrict__ n_gb,
    const float* __restrict__ p_g,  const float* __restrict__ p_b) {
  extern __shared__ char smem[];
  __bf16* sRe = (__bf16*)smem;          // 68 x 128 (2-halo each side)
  __bf16* sIm = sRe + 68 * 128;         // 68 x 128
  __bf16* sT0 = sIm + 68 * 128;         // 64 x 256 scratch (conv out, then h1)
  __bf16* sT1 = sT0 + 64 * 256;         // 64 x 256 scratch (t1, then h)
  float*  sSr = (float*)(sT1 + 64 * 256); // 64 x 128 S_re (then pn_re)
  float*  sSi = sSr + 64 * 128;           // 64 x 128 S_im (then pn_im)
  float*  sTh = sSi + 64 * 128;           // 64 x 128 theta
  float*  sG  = sTh + 64 * 128;           // 64 x 128 gate

  const int b  = (int)blockIdx.x >> 4;
  const int x0 = ((int)blockIdx.x & 15) << 6;
  const int tid = (int)threadIdx.x;
  const float* inRe = pRe + (size_t)b * cX * cH;
  const float* inIm = pIm + (size_t)b * cX * cH;

  // stage 0: stage psi (+halo, zero-padded) into LDS as bf16
  for (int idx = tid; idx < 68 * 128; idx += 256) {
    const int r = idx >> 7, c = idx & 127;
    const int x = x0 + r - 2;
    float vr = 0.f, vi = 0.f;
    if (x >= 0 && x < cX) { vr = inRe[(size_t)x * cH + c]; vi = inIm[(size_t)x * cH + c]; }
    sRe[idx] = (__bf16)vr; sIm[idx] = (__bf16)vi;
  }
  __syncthreads();

  // stage 1: conv (5 shifted HxH GEMMs) + bias -> sT0 [re | im]
  {
    Seg cs[cK];
    for (int d = 0; d < cK; ++d) cs[d] = { sRe + d * 128, 128, convR + (size_t)d * cH * cH, cH, cH };
    gemm64(cs, cK, cH, EpiBf{ s_rb, sT0, 256, 0 });
    for (int d = 0; d < cK; ++d) cs[d] = { sIm + d * 128, 128, convI + (size_t)d * cH * cH, cH, cH };
    gemm64(cs, cK, cH, EpiBf{ s_ib, sT0 + 128, 256, 0 });
  }
  __syncthreads();
  // stage 2: shared pointwise mix -> S (f32)
  { Seg s = { sT0,       256, mW, cH, cH }; gemm64(&s, 1, cH, EpiF32{ s_mb, sSr, 128, 0 }); }
  { Seg s = { sT0 + 128, 256, mW, cH, cH }; gemm64(&s, 1, cH, EpiF32{ s_mb, sSi, 128, 0 }); }
  __syncthreads();
  // stage 3: rotation MLP: t1 = gelu(feat @ rW1^T + b) -> sT1
  { Seg fs[2] = { { sRe + 2 * 128, 128, rW1,      cH2, cH },
                  { sIm + 2 * 128, 128, rW1 + cH, cH2, cH } };
    gemm64(fs, 2, cH, EpiBf{ r_b1, sT1, 128, 1 }); }
  __syncthreads();
  // theta = tanh(t1 @ rW2^T + b) -> sTh (f32)
  { Seg s = { sT1, 128, rW2, cH, cH }; gemm64(&s, 1, cH, EpiF32{ r_b2, sTh, 128, 2 }); }
  __syncthreads();
  // stage 4: gated MLP: h1 = gelu(feat @ nW1^T + b) -> sT0 (overwrites conv tmp)
  { Seg fs[2] = { { sRe + 2 * 128, 128, nW1,      cH2, cH },
                  { sIm + 2 * 128, 128, nW1 + cH, cH2, cH } };
    gemm64(fs, 2, cH2, EpiBf{ n_b1, sT0, 256, 1 }); }
  __syncthreads();
  // h = h1 @ nW2^T + b -> sT1 ; g = sigmoid(feat @ nGw^T + b) -> sG
  { Seg s = { sT0, 256, nW2, cH2, cH2 }; gemm64(&s, 1, cH2, EpiBf{ n_b2, sT1, 256, 0 }); }
  { Seg fs[2] = { { sRe + 2 * 128, 128, nGw,      cH2, cH },
                  { sIm + 2 * 128, 128, nGw + cH, cH2, cH } };
    gemm64(fs, 2, cH, EpiF32{ n_gb, sG, 128, 3 }); }
  __syncthreads();

  // stage 5: combine S + R + N, Euler step, clip, LayerNorm over flat [H,2].
  // 4 threads per row (always same wave32) + shuffle reduction.
  {
    const int m = tid >> 2;          // row 0..63
    const int p = tid & 3;           // channel quarter (32 each)
    const int x = x0 + m;
    const float* rRe = inRe + (size_t)x * cH;
    const float* rIm = inIm + (size_t)x * cH;
    float sum = 0.f, sq = 0.f;
    for (int n = p * 32; n < p * 32 + 32; ++n) {
      const float reo = rRe[n], imo = rIm[n];
      const float th = sTh[m * 128 + n], gg = sG[m * 128 + n];
      const float hr = (float)sT1[m * 256 + n];
      const float hi = (float)sT1[m * 256 + 128 + n];
      float dr = sSr[m * 128 + n] - th * imo + gg * hr;
      float di = sSi[m * 128 + n] + th * reo + gg * hi;
      dr = fminf(5.f, fmaxf(-5.f, dr));
      di = fminf(5.f, fmaxf(-5.f, di));
      const float pr = fminf(10.f, fmaxf(-10.f, reo + cDT * dr));
      const float pi = fminf(10.f, fmaxf(-10.f, imo + cDT * di));
      sSr[m * 128 + n] = pr; sSi[m * 128 + n] = pi;
      sum += pr + pi; sq += pr * pr + pi * pi;
    }
    sum += __shfl_xor(sum, 1);  sq += __shfl_xor(sq, 1);
    sum += __shfl_xor(sum, 2);  sq += __shfl_xor(sq, 2);
    const float mu  = sum * (1.f / 256.f);
    const float var = sq * (1.f / 256.f) - mu * mu;   // ddof=0
    const float inv = rsqrtf(var + 1e-5f);
    float* wRe = oRe + ((size_t)b * cX + x) * cH;
    float* wIm = oIm + ((size_t)b * cX + x) * cH;
    for (int n = p * 32; n < p * 32 + 32; ++n) {
      wRe[n] = (sSr[m * 128 + n] - mu) * inv * p_g[2 * n]     + p_b[2 * n];      // flat[2d]   = re
      wIm[n] = (sSi[m * 128 + n] - mu) * inv * p_g[2 * n + 1] + p_b[2 * n + 1];  // flat[2d+1] = im
    }
  }
}

// ---------------- decoder ----------------
__global__ __launch_bounds__(256) void dec_kernel(
    const float* __restrict__ pRe, const float* __restrict__ pIm,
    const __bf16* __restrict__ w1, const float* __restrict__ b1,
    const float* __restrict__ w2, const float* __restrict__ b2,
    float* __restrict__ out, int tstep) {
  extern __shared__ char smem[];
  __bf16* sF = (__bf16*)smem;            // 64 x 256 feat
  float*  sH = (float*)(sF + 64 * 256);  // 64 x 128 hidden
  const int b  = (int)blockIdx.x >> 4;
  const int x0 = ((int)blockIdx.x & 15) << 6;
  const int tid = (int)threadIdx.x;
  const float* inRe = pRe + ((size_t)b * cX + x0) * cH;
  const float* inIm = pIm + ((size_t)b * cX + x0) * cH;
  for (int idx = tid; idx < 64 * 128; idx += 256) {
    const int r = idx >> 7, c = idx & 127;
    sF[r * 256 + c]       = (__bf16)inRe[(size_t)r * cH + c];
    sF[r * 256 + 128 + c] = (__bf16)inIm[(size_t)r * cH + c];
  }
  __syncthreads();
  { Seg s = { sF, 256, w1, cH2, cH2 }; gemm64(&s, 1, cH, EpiF32{ b1, sH, 128, 1 }); }
  __syncthreads();
  // final H->1 dot: 4 threads per row + shuffle reduction
  {
    const int m = tid >> 2;
    const int p = tid & 3;
    float acc = 0.f;
    for (int n = p * 32; n < p * 32 + 32; ++n) acc += sH[m * 128 + n] * w2[n];
    acc += __shfl_xor(acc, 1);
    acc += __shfl_xor(acc, 2);
    if (p == 0)
      out[((size_t)b * cTOUT + tstep) * cX + x0 + m] = acc + b2[0];
  }
}

// ---------------- host ----------------
extern "C" void kernel_launch(void* const* d_in, const int* in_sizes, int n_in,
                              void* d_out, int out_size, void* d_ws, size_t ws_size,
                              hipStream_t stream) {
  (void)in_sizes; (void)n_in; (void)out_size; (void)ws_size;
  char* ws = (char*)d_ws;
  size_t off = 0;
  auto take = [&](size_t bytes) -> char* {
    char* p = ws + off; off = (off + bytes + 255) & ~(size_t)255; return p;
  };
  char* encW1 = take(8192u   * 2);   // [128][64]
  char* encW2 = take(32768u  * 2);   // [256][128]
  char* convR = take(245760u * 2);   // [3][5][128][128]
  char* convI = take(245760u * 2);
  char* mW    = take(49152u  * 2);   // [3][128][128]
  char* rW1   = take(98304u  * 2);   // [3][128][256]
  char* rW2   = take(49152u  * 2);
  char* nW1   = take(196608u * 2);   // [3][256][256]
  char* nW2   = take(196608u * 2);
  char* nGw   = take(98304u  * 2);   // [3][128][256]
  char* decW1 = take(32768u  * 2);   // [128][256]
  float* reA = (float*)take(1048576u * 4);   // [B][X][H]
  float* imA = (float*)take(1048576u * 4);
  float* reB = (float*)take(1048576u * 4);
  float* imB = (float*)take(1048576u * 4);

  auto cvt = [&](const void* src, char* dst, int n) {
    cvt_kernel<<<(n + 255) / 256, 256, 0, stream>>>((const float*)src, (__bf16*)dst, n);
  };
  cvt(d_in[1],  encW1, 8192);
  cvt(d_in[3],  encW2, 32768);
  conv_repack_kernel<<<(245760 + 255) / 256, 256, 0, stream>>>((const float*)d_in[5], (__bf16*)convR, 245760);
  conv_repack_kernel<<<(245760 + 255) / 256, 256, 0, stream>>>((const float*)d_in[7], (__bf16*)convI, 245760);
  cvt(d_in[9],  mW,   49152);
  cvt(d_in[11], rW1,  98304);
  cvt(d_in[13], rW2,  49152);
  cvt(d_in[15], nW1,  196608);
  cvt(d_in[17], nW2,  196608);
  cvt(d_in[19], nGw,  98304);
  cvt(d_in[23], decW1, 32768);

  const size_t encLds = 64 * 64 * 2 + 64 * 128 * 2;                        // 24 KB
  const size_t blkLds = 2 * (68 * 128 * 2) + 2 * (64 * 256 * 2) + 4 * (64 * 128 * 4); // ~226 KB (< 320 KB WGP LDS)
  const size_t decLds = 64 * 256 * 2 + 64 * 128 * 4;                       // 64 KB

  enc_kernel<<<cB * (cX / 64), 256, encLds, stream>>>(
      (const float*)d_in[0], (const __bf16*)encW1, (const float*)d_in[2],
      (const __bf16*)encW2, (const float*)d_in[4], reA, imA);

  float* curRe = reA; float* curIm = imA; float* nxtRe = reB; float* nxtIm = imB;
  for (int t = 0; t < cTOUT; ++t) {
    for (int i = 0; i < cDEPTH; ++i) {
      block_kernel<<<cB * (cX / 64), 256, blkLds, stream>>>(
          curRe, curIm, nxtRe, nxtIm,
          (const __bf16*)(convR + (size_t)i * cK * cH * cH * 2),
          (const __bf16*)(convI + (size_t)i * cK * cH * cH * 2),
          (const __bf16*)(mW  + (size_t)i * cH * cH * 2),
          (const __bf16*)(rW1 + (size_t)i * cH * cH2 * 2),
          (const __bf16*)(rW2 + (size_t)i * cH * cH * 2),
          (const __bf16*)(nW1 + (size_t)i * cH2 * cH2 * 2),
          (const __bf16*)(nW2 + (size_t)i * cH2 * cH2 * 2),
          (const __bf16*)(nGw + (size_t)i * cH * cH2 * 2),
          (const float*)d_in[6]  + i * cH,  (const float*)d_in[8]  + i * cH,
          (const float*)d_in[10] + i * cH,  (const float*)d_in[12] + i * cH,
          (const float*)d_in[14] + i * cH,  (const float*)d_in[16] + i * cH2,
          (const float*)d_in[18] + i * cH2, (const float*)d_in[20] + i * cH,
          (const float*)d_in[21] + i * cH2, (const float*)d_in[22] + i * cH2);
      float* tr = curRe; curRe = nxtRe; nxtRe = tr;
      tr = curIm; curIm = nxtIm; nxtIm = tr;
    }
    dec_kernel<<<cB * (cX / 64), 256, decLds, stream>>>(
        curRe, curIm, (const __bf16*)decW1, (const float*)d_in[24],
        (const float*)d_in[25], (const float*)d_in[26], (float*)d_out, t);
  }
}
